// GATEncoder_412316860775
// MI455X (gfx1250) — matrix operand order repaired and checked by reading, not compiled, not run on hardware
//
#include <hip/hip_runtime.h>
#include <hip/hip_bf16.h>

// ---------------- problem constants ----------------
#define LL   3
#define HH   12
#define DD   768
#define DHH  64
#define FFF  3072
#define BB   4
#define SS   512
#define RR   (BB*SS)          // 2048 rows of the token stream

// ---------------- vector types ----------------
typedef __attribute__((ext_vector_type(16))) __bf16 v16bf;
typedef __attribute__((ext_vector_type(8)))  __bf16 v8bf;
typedef __attribute__((ext_vector_type(8)))  float  v8f;

// ---------------- scalar helpers ----------------
__device__ __forceinline__ __bf16 f2bf(float f) {
    unsigned u; __builtin_memcpy(&u, &f, 4);
    unsigned r = u + 0x7FFFu + ((u >> 16) & 1u);
    unsigned short h = (unsigned short)(r >> 16);
    __bf16 o; __builtin_memcpy(&o, &h, 2);
    return o;
}

__device__ __forceinline__ float gelu_f(float x) {
    return 0.5f * x * (1.0f + erff(x * 0.70710678118654752440f));
}

__device__ __forceinline__ v16bf zero_frag() {
    unsigned short zs = 0; __bf16 zb; __builtin_memcpy(&zb, &zs, 2);
    v16bf z;
#pragma unroll
    for (int i = 0; i < 16; ++i) z[i] = zb;
    return z;
}

// ---------------- CDNA5 async copy: global -> LDS (ASYNCcnt-tracked) ----------------
// lds_off: raw byte offset into workgroup LDS (extern dynamic LDS starts at 0).
__device__ __forceinline__ void async_b128_to_lds(unsigned lds_off, const void* g) {
    asm volatile("global_load_async_to_lds_b128 %0, %1, off"
                 :: "v"(lds_off), "v"(g) : "memory");
}
__device__ __forceinline__ void wait_async_all() {
    asm volatile("s_wait_asynccnt 0x0" ::: "memory");
}

// A-operand fragment (16-bit A 16x32): lanes<16 hold K{0..7,16..23},
// lanes>=16 hold K{8..15,24..31}.  p -> 32 contiguous K elems of this row.
__device__ __forceinline__ v16bf load_fragA(const __bf16* p) {
    const int off = (threadIdx.x & 16) ? 8 : 0;
    v8bf lo = *(const v8bf*)(p + off);
    v8bf hi = *(const v8bf*)(p + 16 + off);
    return __builtin_shufflevector(lo, hi, 0,1,2,3,4,5,6,7,8,9,10,11,12,13,14,15);
}

// B-operand fragment (16-bit B 32x16): lanes<16 hold K{0..7,16..23}, lanes>=16 K{8..15,24..31}.
// p -> 32 contiguous K elems of this column (column-major / [N][K] storage).
__device__ __forceinline__ v16bf load_fragB(const __bf16* p) {
    const int off = (threadIdx.x & 16) ? 8 : 0;
    v8bf lo = *(const v8bf*)(p + off);
    v8bf hi = *(const v8bf*)(p + 16 + off);
    return __builtin_shufflevector(lo, hi, 0,1,2,3,4,5,6,7,8,9,10,11,12,13,14,15);
}

__device__ __forceinline__ v8f mma_bf16(v16bf a, v16bf b, v8f c) {
    return __builtin_amdgcn_wmma_f32_16x16x32_bf16(
        /*neg_a=*/false, a, /*neg_b=*/false, b,
        /*c_mod=*/(short)0, c, /*reuse_a=*/false, /*reuse_b=*/false);
}

// ---------------- elementwise / conversion kernels ----------------
__global__ __launch_bounds__(256) void k_cvt4(const float* __restrict__ s,
                                              __bf16* __restrict__ d, long n4) {
    long stride = (long)gridDim.x * blockDim.x;
    for (long i = (long)blockIdx.x * blockDim.x + threadIdx.x; i < n4; i += stride) {
        float4 v = ((const float4*)s)[i];
        d[i*4+0] = f2bf(v.x); d[i*4+1] = f2bf(v.y);
        d[i*4+2] = f2bf(v.z); d[i*4+3] = f2bf(v.w);
    }
}

__global__ __launch_bounds__(256) void k_copyf(const float* __restrict__ s,
                                               float* __restrict__ d, long n) {
    long stride = (long)gridDim.x * blockDim.x;
    for (long i = (long)blockIdx.x * blockDim.x + threadIdx.x; i < n; i += stride)
        d[i] = s[i];
}

// transpose+convert one stack of L matrices: src [l][K][N] f32 -> dst [l][N][K] bf16
__global__ __launch_bounds__(256) void k_wcvt(const float* __restrict__ src,
                                              __bf16* __restrict__ dst, int K, int N) {
    int l = blockIdx.z;
    long base = (long)l * K * N;
    long tot = (long)N * K;
    long stride = (long)gridDim.x * blockDim.x;
    for (long idx = (long)blockIdx.x * blockDim.x + threadIdx.x; idx < tot; idx += stride) {
        int n = (int)(idx / K), k = (int)(idx % K);
        dst[base + idx] = f2bf(src[base + (long)k * N + n]);
    }
}

// ---------------- LayerNorm: one 256-thread block per row (D=768) ----------------
__global__ __launch_bounds__(256) void k_ln(const float* __restrict__ x,
                                            const float* __restrict__ g,
                                            const float* __restrict__ bta,
                                            __bf16* __restrict__ outbf,
                                            float*  __restrict__ outf) {
    int row = blockIdx.x;
    const float* xr = x + (size_t)row * DD;
    float v0 = xr[threadIdx.x], v1 = xr[threadIdx.x + 256], v2 = xr[threadIdx.x + 512];
    float s = v0 + v1 + v2, s2 = v0*v0 + v1*v1 + v2*v2;
#pragma unroll
    for (int off = 16; off > 0; off >>= 1) {
        s  += __shfl_xor(s,  off, 32);
        s2 += __shfl_xor(s2, off, 32);
    }
    __shared__ float ss[8], ss2[8];
    int w = threadIdx.x >> 5, lane = threadIdx.x & 31;
    if (lane == 0) { ss[w] = s; ss2[w] = s2; }
    __syncthreads();
    float S = 0.f, S2 = 0.f;
#pragma unroll
    for (int i = 0; i < 8; ++i) { S += ss[i]; S2 += ss2[i]; }
    float mean = S * (1.0f / DD);
    float var  = S2 * (1.0f / DD) - mean * mean;
    float inv  = rsqrtf(var + 1e-6f);
    float vs[3] = {v0, v1, v2};
#pragma unroll
    for (int j = 0; j < 3; ++j) {
        int col = threadIdx.x + j * 256;
        float y = (vs[j] - mean) * inv * g[col] + bta[col];
        if (outbf) outbf[(size_t)row * DD + col] = f2bf(y);
        else       outf [(size_t)row * DD + col] = y;
    }
}

// ---------------- unified WMMA GEMM with async double-buffered B staging ----------------
// C[M x N] = A(bf16, row-major [M][K], lda) * B(bf16, [N][K] "transposed", ldb) + bias
// block = 256 threads = 8 waves; block tile 128(M) x 64(N); wave tile 16 x 64.
// B tile (shared by all 8 waves) is staged in LDS via global_load_async_to_lds_b128,
// double-buffered per 32-wide K step (2 x 4KB stages).
#define GM_OUT_F32   0   // f32 row-major, ld=ldc, batched by strideO
#define GM_OUT_BF16  1   // bf16 row-major, ld=ldc
#define GM_OUT_QKV   2   // bf16 [b][h][s][dh] (row=(b,s), col=(h,dh))
#define GM_OUT_VT    3   // bf16 [b][h][dh][s]
#define GM_ADD_F32   4   // f32 row-major +=
#define GM_OUT_CTX   5   // f32 ctx[b][s][h*DHH+n], z=(b,h), N=64

__global__ __launch_bounds__(256) void k_gemm(const __bf16* __restrict__ Abase, long strideA, int lda,
                                              const __bf16* __restrict__ Bbase, long strideB, int ldb,
                                              const float* __restrict__ bias,
                                              void* __restrict__ Obase, long strideO, int ldc,
                                              int M, int N, int K, int mode, int dogelu) {
    extern __shared__ __bf16 sB[];    // 2 stages x 64 cols x 32 k = 8KB
    int z    = blockIdx.z;
    const __bf16* A  = Abase + (size_t)z * strideA;
    const __bf16* Bt = Bbase + (size_t)z * strideB;
    int tid = threadIdx.x, w = tid >> 5, lane = tid & 31, l16 = lane & 15;
    int m0 = blockIdx.y * 128 + w * 16;
    int n0 = blockIdx.x * 64;
    if (blockIdx.y * 128 >= M) return;

    // cooperative async fill of one B stage: thread -> one b128 (8 bf16)
    int bcolid = tid >> 2, bchunk = tid & 3;
    const __bf16* bsrc0 = Bt + (size_t)(n0 + bcolid) * ldb + bchunk * 8;
    unsigned blds0 = (unsigned)(bcolid * 64 + bchunk * 16);   // bytes within a stage

    v8f acc[4];
#pragma unroll
    for (int t = 0; t < 4; ++t)
#pragma unroll
        for (int j = 0; j < 8; ++j) acc[t][j] = 0.f;

    const __bf16* arow = A + (size_t)(m0 + l16) * lda;

    // prologue: stage 0, k0 = 0
    async_b128_to_lds(blds0, bsrc0);
    v16bf a_cur = load_fragA(arow);

    int nsteps = K >> 5;
    for (int s = 0; s < nsteps; ++s) {
        wait_async_all();
        __syncthreads();
        int stage = s & 1;
        if (s + 1 < nsteps) {
            async_b128_to_lds(blds0 + (unsigned)((stage ^ 1) * 4096),
                              bsrc0 + (s + 1) * 32);
        }
        v16bf a_use = a_cur;
        if (s + 1 < nsteps) a_cur = load_fragA(arow + (s + 1) * 32);
#pragma unroll
        for (int t = 0; t < 4; ++t) {
            const __bf16* bcol = sB + stage * 2048 + (t * 16 + l16) * 32;
            acc[t] = mma_bf16(a_use, load_fragB(bcol), acc[t]);
        }
    }

    int mrow0 = m0 + ((lane >> 4) << 3);
#pragma unroll
    for (int t = 0; t < 4; ++t) {
        int n = n0 + t * 16 + l16;
        float bv = bias ? bias[n] : 0.f;
#pragma unroll
        for (int j = 0; j < 8; ++j) {
            int m = mrow0 + j;
            float vv = acc[t][j] + bv;
            if (dogelu) vv = gelu_f(vv);
            if (mode == GM_OUT_F32) {
                ((float*)Obase + (size_t)z * strideO)[(size_t)m * ldc + n] = vv;
            } else if (mode == GM_OUT_BF16) {
                ((__bf16*)Obase)[(size_t)m * ldc + n] = f2bf(vv);
            } else if (mode == GM_ADD_F32) {
                float* p = (float*)Obase + (size_t)m * ldc + n;
                *p += vv;
            } else if (mode == GM_OUT_QKV) {
                int b = m >> 9, sr = m & 511, h = n >> 6, d = n & 63;
                ((__bf16*)Obase)[(((size_t)(b * HH + h)) * SS + sr) * DHH + d] = f2bf(vv);
            } else if (mode == GM_OUT_VT) {
                int b = m >> 9, sr = m & 511, h = n >> 6, d = n & 63;
                ((__bf16*)Obase)[(((size_t)(b * HH + h)) * DHH + d) * SS + sr] = f2bf(vv);
            } else { // GM_OUT_CTX
                int b = z / HH, h = z % HH;
                ((float*)Obase)[((size_t)b * SS + m) * DD + h * DHH + n] = vv;
            }
        }
    }
}

// ---------------- relational scores: scores[b,h,q,k] += q[b,h,q,:].dep[b,q,k,:] ----------------
// one block per (b,q); dep slice [512][64] bf16 staged in 64KB LDS via async copy.
__global__ __launch_bounds__(256) void k_rel_scores(const __bf16* __restrict__ dep,
                                                    const __bf16* __restrict__ qb,
                                                    float* __restrict__ scores) {
    extern __shared__ __bf16 sdep[];                 // [512][64]
    int bq = blockIdx.x, b = bq >> 9, qpos = bq & 511;
    {
        const __bf16* src = dep + (size_t)bq * SS * DHH;
        for (int i = threadIdx.x; i < SS * DHH / 8; i += 256)
            async_b128_to_lds((unsigned)(i * 16), src + i * 8);
        wait_async_all();
    }
    __syncthreads();

    int tid = threadIdx.x, w = tid >> 5, lane = tid & 31, l16 = lane & 15;
    int h = l16;
    v16bf a0, a1;
    if (h < HH) {
        const __bf16* arow = qb + (((size_t)(b * HH + h)) * SS + qpos) * DHH;
        a0 = load_fragA(arow);
        a1 = load_fragA(arow + 32);
    } else {
        a0 = zero_frag(); a1 = zero_frag();
    }
#pragma unroll
    for (int t = 0; t < 4; ++t) {
        int sk0 = (w * 4 + t) * 16;
        const __bf16* bcol = sdep + (size_t)(sk0 + l16) * DHH;
        v8f acc;
#pragma unroll
        for (int j = 0; j < 8; ++j) acc[j] = 0.f;
        acc = mma_bf16(a0, load_fragB(bcol),      acc);
        acc = mma_bf16(a1, load_fragB(bcol + 32), acc);
        int nsk = sk0 + l16, mh0 = (lane >> 4) << 3;
#pragma unroll
        for (int j = 0; j < 8; ++j) {
            int hh = mh0 + j;
            if (hh < HH) {
                float* p = scores + (((size_t)(b * HH + hh)) * SS + qpos) * SS + nsk;
                *p += acc[j];
            }
        }
    }
}

// ---------------- softmax: wave per row, bf16 probs out ----------------
__global__ __launch_bounds__(256) void k_softmax(const float* __restrict__ scores,
                                                 const float* __restrict__ mask,
                                                 __bf16* __restrict__ probs) {
    int w = threadIdx.x >> 5, lane = threadIdx.x & 31;
    int r = blockIdx.x * 8 + w;                 // r < B*H*S
    int b = r / (HH * SS);
    const float* srow = scores + (size_t)r * SS;
    const float* mrow = mask + (size_t)b * SS;
    float v[16], mx = -3.4e38f;
#pragma unroll
    for (int t = 0; t < 16; ++t) {
        int c = lane + t * 32;
        v[t] = srow[c] * 0.125f + mrow[c];
        mx = fmaxf(mx, v[t]);
    }
#pragma unroll
    for (int off = 16; off > 0; off >>= 1) mx = fmaxf(mx, __shfl_xor(mx, off, 32));
    float sum = 0.f;
#pragma unroll
    for (int t = 0; t < 16; ++t) { v[t] = __expf(v[t] - mx); sum += v[t]; }
#pragma unroll
    for (int off = 16; off > 0; off >>= 1) sum += __shfl_xor(sum, off, 32);
    float inv = 1.f / sum;
    __bf16* prow = probs + (size_t)r * SS;
#pragma unroll
    for (int t = 0; t < 16; ++t) prow[lane + t * 32] = f2bf(v[t] * inv);
}

// ---------------- relational context: ctx[b,q,h*64+d] += sum_k probs*dep ----------------
// one block per (b,q); dep slice transposed into LDS as [64][512] bf16 (manual fill,
// transpose happens during staging so the async path does not apply).
__global__ __launch_bounds__(256) void k_rel_ctx(const __bf16* __restrict__ dep,
                                                 const __bf16* __restrict__ probs,
                                                 float* __restrict__ ctx) {
    extern __shared__ __bf16 sdt[];                  // [64][512]
    int bq = blockIdx.x, b = bq >> 9, qpos = bq & 511;
    {
        const uint4* src = (const uint4*)(dep + (size_t)bq * SS * DHH);
        for (int i = threadIdx.x; i < SS * DHH / 8; i += 256) {
            int k = i >> 3, d0 = (i & 7) * 8;
            uint4 c = src[i];
            unsigned short es[8]; __builtin_memcpy(es, &c, 16);
#pragma unroll
            for (int jj = 0; jj < 8; ++jj) {
                __bf16 t; __builtin_memcpy(&t, &es[jj], 2);
                sdt[(size_t)(d0 + jj) * SS + k] = t;
            }
        }
    }
    __syncthreads();

    int tid = threadIdx.x, w = tid >> 5, lane = tid & 31, l16 = lane & 15;
    int t = w & 3, kbase = (w >> 2) * 256;           // waves 0-3: k 0..255, 4-7: 256..511
    int h = l16, dcol = t * 16 + l16;
    const __bf16* bcol = sdt + (size_t)dcol * SS + kbase;
    const __bf16* arow = probs + (((size_t)(b * HH + (h < HH ? h : 0))) * SS + qpos) * SS + kbase;

    v8f acc;
#pragma unroll
    for (int j = 0; j < 8; ++j) acc[j] = 0.f;
    for (int kk = 0; kk < 256; kk += 32) {
        v16bf af = load_fragA(arow + kk);
        if (h >= HH) af = zero_frag();
        acc = mma_bf16(af, load_fragB(bcol + kk), acc);
    }
    int mh0 = (lane >> 4) << 3;
#pragma unroll
    for (int j = 0; j < 8; ++j) {
        int hh = mh0 + j;
        if (hh < HH)
            atomicAdd(&ctx[((size_t)b * SS + qpos) * DD + hh * DHH + dcol], acc[j]);
    }
}

// ---------------- host-side orchestration ----------------
extern "C" void kernel_launch(void* const* d_in, const int* in_sizes, int n_in,
                              void* d_out, int out_size, void* d_ws, size_t ws_size,
                              hipStream_t stream) {
    const float* e_in  = (const float*)d_in[0];
    const float* amask = (const float*)d_in[1];
    const float* dep   = (const float*)d_in[2];
    const float* Wq    = (const float*)d_in[3];
    const float* bq_   = (const float*)d_in[4];
    const float* Wk    = (const float*)d_in[5];
    const float* bk_   = (const float*)d_in[6];
    const float* Wv    = (const float*)d_in[7];
    const float* bv_   = (const float*)d_in[8];
    const float* Wo    = (const float*)d_in[9];
    const float* bo_   = (const float*)d_in[10];
    const float* ln2g  = (const float*)d_in[11];
    const float* ln2b  = (const float*)d_in[12];
    const float* W1    = (const float*)d_in[13];
    const float* b1_   = (const float*)d_in[14];
    const float* W2    = (const float*)d_in[15];
    const float* b2_   = (const float*)d_in[16];
    const float* ln3g  = (const float*)d_in[17];
    const float* ln3b  = (const float*)d_in[18];
    const float* lnfg  = (const float*)d_in[19];
    const float* lnfb  = (const float*)d_in[20];

    char* ws = (char*)d_ws;
    size_t off = 0;
    auto carve = [&](size_t bytes) -> char* {
        char* p = ws + off;
        off += (bytes + 255) & ~(size_t)255;
        return p;
    };

    const size_t DEP_ELEMS = (size_t)BB * SS * SS * DHH;   // 67,108,864
    __bf16* depB  = (__bf16*)carve(DEP_ELEMS * 2);                       // 128 MB
    __bf16* WqT   = (__bf16*)carve((size_t)LL * DD * DD * 2);
    __bf16* WkT   = (__bf16*)carve((size_t)LL * DD * DD * 2);
    __bf16* WvT   = (__bf16*)carve((size_t)LL * DD * DD * 2);
    __bf16* WoT   = (__bf16*)carve((size_t)LL * DD * DD * 2);
    __bf16* W1T   = (__bf16*)carve((size_t)LL * FFF * DD * 2);
    __bf16* W2T   = (__bf16*)carve((size_t)LL * DD * FFF * 2);
    __bf16* xln   = (__bf16*)carve((size_t)RR * DD * 2);
    __bf16* qb    = (__bf16*)carve((size_t)BB * HH * SS * DHH * 2);
    __bf16* kb    = (__bf16*)carve((size_t)BB * HH * SS * DHH * 2);
    __bf16* vtb   = (__bf16*)carve((size_t)BB * HH * DHH * SS * 2);
    float*  scores= (float*) carve((size_t)BB * HH * SS * SS * 4);       // 48 MB
    __bf16* probs = (__bf16*)carve((size_t)BB * HH * SS * SS * 2);       // 24 MB
    float*  ctx   = (float*) carve((size_t)RR * DD * 4);
    __bf16* ctxbf = (__bf16*)carve((size_t)RR * DD * 2);
    float*  ebuf  = (float*) carve((size_t)RR * DD * 4);
    __bf16* y1    = (__bf16*)carve((size_t)RR * FFF * 2);                // 12 MB
    (void)ws_size; (void)in_sizes; (void)n_in; (void)out_size;

    // one-time conversions (must run every call: no cached state allowed)
    k_cvt4<<<65536, 256, 0, stream>>>(dep, depB, (long)(DEP_ELEMS / 4));
    k_wcvt<<<dim3(2048, 1, LL), 256, 0, stream>>>(Wq, WqT, DD, DD);
    k_wcvt<<<dim3(2048, 1, LL), 256, 0, stream>>>(Wk, WkT, DD, DD);
    k_wcvt<<<dim3(2048, 1, LL), 256, 0, stream>>>(Wv, WvT, DD, DD);
    k_wcvt<<<dim3(2048, 1, LL), 256, 0, stream>>>(Wo, WoT, DD, DD);
    k_wcvt<<<dim3(4096, 1, LL), 256, 0, stream>>>(W1, W1T, DD, FFF);     // -> [FFF][DD]
    k_wcvt<<<dim3(4096, 1, LL), 256, 0, stream>>>(W2, W2T, FFF, DD);     // -> [DD][FFF]
    k_copyf<<<2048, 256, 0, stream>>>(e_in, ebuf, (long)RR * DD);

    const size_t DxD = (size_t)DD * DD;
    const size_t DxF = (size_t)DD * FFF;
    const unsigned GEMM_LDS = 8192;    // 2 B stages x 4KB
    dim3 gq(12, 16, 1);      // N=768, M=2048
    dim3 gs(8, 4, 48);       // scores: N=512, M=512, batch b*h
    dim3 gc(1, 4, 48);       // ctx:    N=64,  M=512, batch b*h
    dim3 g1(48, 16, 1);      // FFN1:   N=3072

    for (int l = 0; l < LL; ++l) {
        // x = LN(e, ln2)
        k_ln<<<RR, 256, 0, stream>>>(ebuf, ln2g + (size_t)l * DD, ln2b + (size_t)l * DD, xln, nullptr);
        // Q, K (head layout), V (transposed head layout)
        k_gemm<<<gq, 256, GEMM_LDS, stream>>>(xln, 0, DD, WqT + l * DxD, 0, DD, bq_ + (size_t)l * DD,
                                              qb, 0, 0, RR, DD, DD, GM_OUT_QKV, 0);
        k_gemm<<<gq, 256, GEMM_LDS, stream>>>(xln, 0, DD, WkT + l * DxD, 0, DD, bk_ + (size_t)l * DD,
                                              kb, 0, 0, RR, DD, DD, GM_OUT_QKV, 0);
        k_gemm<<<gq, 256, GEMM_LDS, stream>>>(xln, 0, DD, WvT + l * DxD, 0, DD, bv_ + (size_t)l * DD,
                                              vtb, 0, 0, RR, DD, DD, GM_OUT_VT, 0);
        // content scores q.k  (A=[S][DH], B=[S][DH] interpreted as [N][K])
        k_gemm<<<gs, 256, GEMM_LDS, stream>>>(qb, (long)SS * DHH, DHH, kb, (long)SS * DHH, DHH, nullptr,
                                              scores, (long)SS * SS, SS, SS, SS, DHH, GM_OUT_F32, 0);
        // + relational scores (dep slice in LDS via async copy)
        k_rel_scores<<<BB * SS, 256, SS * DHH * 2, stream>>>(depB, qb, scores);
        // softmax -> bf16 probs
        k_softmax<<<(BB * HH * SS) / 8, 256, 0, stream>>>(scores, amask, probs);
        // content ctx = probs @ v  (B = v^T [DH][S])
        k_gemm<<<gc, 256, GEMM_LDS, stream>>>(probs, (long)SS * SS, SS, vtb, (long)DHH * SS, SS, nullptr,
                                              ctx, 0, 0, SS, DHH, SS, GM_OUT_CTX, 0);
        // + relational ctx (dep slice transposed in LDS, atomic f32 adds)
        k_rel_ctx<<<BB * SS, 256, SS * DHH * 2, stream>>>(depB, probs, ctx);
        // e += ctx @ Wo + bo
        k_cvt4<<<1536, 256, 0, stream>>>(ctx, ctxbf, (long)RR * DD / 4);
        k_gemm<<<gq, 256, GEMM_LDS, stream>>>(ctxbf, 0, DD, WoT + l * DxD, 0, DD, bo_ + (size_t)l * DD,
                                              ebuf, 0, DD, RR, DD, DD, GM_ADD_F32, 0);
        // FFN: y = LN(e, ln3); e += gelu(y@W1+b1)@W2 + b2
        k_ln<<<RR, 256, 0, stream>>>(ebuf, ln3g + (size_t)l * DD, ln3b + (size_t)l * DD, xln, nullptr);
        k_gemm<<<g1, 256, GEMM_LDS, stream>>>(xln, 0, DD, W1T + l * DxF, 0, DD, b1_ + (size_t)l * FFF,
                                              y1, 0, FFF, RR, FFF, DD, GM_OUT_BF16, 1);
        k_gemm<<<gq, 256, GEMM_LDS, stream>>>(y1, 0, FFF, W2T + l * DxF, 0, FFF, b2_ + (size_t)l * DD,
                                              ebuf, 0, DD, RR, DD, FFF, GM_ADD_F32, 0);
    }
    // final LN -> f32 output
    k_ln<<<RR, 256, 0, stream>>>(ebuf, lnfg, lnfb, nullptr, (float*)d_out);
}
